// MoERouter_80169859547410
// MI455X (gfx1250) — matrix operand order repaired and checked by reading, not compiled, not run on hardware
//
#include <hip/hip_runtime.h>
#include <hip/hip_bf16.h>

#define NUM_TOKENS  16384
#define INPUT_DIM   4096
#define NUM_EXPERTS 64
#define TOPK        2

#define BM      64    // tokens per block
#define KSPLIT  2     // K-dimension split across wave groups
#define KCHUNK  (INPUT_DIM / KSPLIT)
#define THREADS 256   // 8 wave32: 4 token-groups x 2 K-splits

typedef __attribute__((ext_vector_type(16))) __bf16 v16bf;
typedef __attribute__((ext_vector_type(8)))  __bf16 bf16x8;
typedef __attribute__((ext_vector_type(4)))  __bf16 bf16x4;
typedef __attribute__((ext_vector_type(8)))  float  v8f;
typedef __attribute__((ext_vector_type(4)))  float  f32x4;

// ---------------------------------------------------------------------------
// Pre-pass: convert W [64, 4096] f32 -> bf16 row-major into d_ws (512 KB).
// W is tiny (1 MB) and L2-resident; converting it once keeps the main loop's
// VALU budget for the token stream only.
// ---------------------------------------------------------------------------
__global__ void wcvt_kernel(const float* __restrict__ W,
                            __bf16* __restrict__ Wb, int n) {
    int base = (blockIdx.x * blockDim.x + threadIdx.x) * 4;
    if (base + 3 < n) {
        f32x4 f = *(const f32x4*)(W + base);
        *(bf16x4*)(Wb + base) = __builtin_convertvector(f, bf16x4);
    }
}

// B-tile load (32x16 bf16, K-consecutive per lane) + WMMA accumulate
#define BTILE(PTR, ACC) do {                                                   \
    bf16x8 _lo = *(const bf16x8*)((PTR) + k);                                  \
    bf16x8 _hi = *(const bf16x8*)((PTR) + k + 16);                             \
    v16bf _b;                                                                  \
    _Pragma("unroll")                                                          \
    for (int _j = 0; _j < 8; ++_j) { _b[_j] = _lo[_j]; _b[8 + _j] = _hi[_j]; } \
    ACC = __builtin_amdgcn_wmma_f32_16x16x32_bf16(                             \
        false, a, false, _b, (short)0, ACC, false, false);                     \
} while (0)

// ---------------------------------------------------------------------------
// Main kernel: per-wave 16-token x 64-expert logits tile via bf16 WMMA,
// K split across wave pairs; LDS reduce; softmax + top-2 epilogue.
// ---------------------------------------------------------------------------
__global__ void __launch_bounds__(THREADS)
router_topk_kernel(const float* __restrict__ tokens,
                   const __bf16* __restrict__ Wb,
                   float* __restrict__ scores_out,
                   int* __restrict__ idx_out) {
    __shared__ float slog[KSPLIT][BM][NUM_EXPERTS + 1];  // +1 pad, 33 KB

    const int tid    = threadIdx.x;
    const int wave   = tid >> 5;
    const int lane   = tid & 31;
    const int laneLo = lane & 15;
    const int laneHi = lane >> 4;          // 0 or 1
    const int group  = wave & 3;           // which 16-token tile
    const int ksp    = wave >> 2;          // which K half

    const int tokBlock = blockIdx.x * BM;
    const int rowA     = tokBlock + group * 16 + laneLo;

    // A-fragment stream: lanes 0-15 take K {0..7,16..23}, lanes 16-31 take
    // K {8..15,24..31} (ISA 16-bit A layout). A lane pair covers one 128B line.
    const float* __restrict__ arow =
        tokens + (size_t)rowA * INPUT_DIM + (laneHi ? 8 : 0);

    // B rows: expert column = 16*n + laneLo; lanes 16-31 start at K+8.
    const int bofs = laneHi ? 8 : 0;
    const __bf16* __restrict__ brow0 = Wb + (size_t)(laneLo +  0) * INPUT_DIM + bofs;
    const __bf16* __restrict__ brow1 = Wb + (size_t)(laneLo + 16) * INPUT_DIM + bofs;
    const __bf16* __restrict__ brow2 = Wb + (size_t)(laneLo + 32) * INPUT_DIM + bofs;
    const __bf16* __restrict__ brow3 = Wb + (size_t)(laneLo + 48) * INPUT_DIM + bofs;

    v8f acc0 = {}, acc1 = {}, acc2 = {}, acc3 = {};

    const int kBeg = ksp * KCHUNK;
    const int kEnd = kBeg + KCHUNK;
    for (int k = kBeg; k < kEnd; k += 32) {
        // ---- A: 16 f32 per lane -> bf16 ----
        f32x4 fa0 = *(const f32x4*)(arow + k);
        f32x4 fa1 = *(const f32x4*)(arow + k + 4);
        f32x4 fa2 = *(const f32x4*)(arow + k + 16);
        f32x4 fa3 = *(const f32x4*)(arow + k + 20);
        __builtin_prefetch(arow + k + 128, 0, 1);  // global_prefetch down the row

        bf16x4 ba0 = __builtin_convertvector(fa0, bf16x4);
        bf16x4 ba1 = __builtin_convertvector(fa1, bf16x4);
        bf16x4 ba2 = __builtin_convertvector(fa2, bf16x4);
        bf16x4 ba3 = __builtin_convertvector(fa3, bf16x4);
        v16bf a;
        #pragma unroll
        for (int j = 0; j < 4; ++j) {
            a[j]      = ba0[j];
            a[4 + j]  = ba1[j];
            a[8 + j]  = ba2[j];
            a[12 + j] = ba3[j];
        }

        // ---- 4 expert tiles: v_wmma_f32_16x16x32_bf16 ----
        BTILE(brow0, acc0);
        BTILE(brow1, acc1);
        BTILE(brow2, acc2);
        BTILE(brow3, acc3);
    }

    // Scatter C (16x16 f32 layout: VGPR r -> M = r + 8*laneHi, N = laneLo)
    #pragma unroll
    for (int r = 0; r < 8; ++r) {
        int tw = group * 16 + r + (laneHi ? 8 : 0);
        slog[ksp][tw][laneLo     ] = acc0[r];
        slog[ksp][tw][laneLo + 16] = acc1[r];
        slog[ksp][tw][laneLo + 32] = acc2[r];
        slog[ksp][tw][laneLo + 48] = acc3[r];
    }
    __syncthreads();

    // ---- Epilogue: one thread per token, softmax + top-2 over 64 experts ----
    if (tid < BM) {
        float l[NUM_EXPERTS];
        float m1 = -3.402823466e38f; int i1 = 0;
        for (int e = 0; e < NUM_EXPERTS; ++e) {
            float v = slog[0][tid][e] + slog[1][tid][e];
            l[e] = v;
            if (v > m1) { m1 = v; i1 = e; }   // strict '>' => lowest index wins ties
        }
        float m2 = -3.402823466e38f; int i2 = 0; float sum = 0.0f;
        for (int e = 0; e < NUM_EXPERTS; ++e) {
            sum += __expf(l[e] - m1);
            if (e != i1 && l[e] > m2) { m2 = l[e]; i2 = e; }
        }
        float inv = 1.0f / sum;
        int gt = tokBlock + tid;
        scores_out[2 * gt    ] = inv;                       // exp(m1-m1)/sum
        scores_out[2 * gt + 1] = __expf(m2 - m1) * inv;
        idx_out[2 * gt    ] = i1;
        idx_out[2 * gt + 1] = i2;
    }
}

extern "C" void kernel_launch(void* const* d_in, const int* in_sizes, int n_in,
                              void* d_out, int out_size, void* d_ws, size_t ws_size,
                              hipStream_t stream) {
    (void)in_sizes; (void)n_in; (void)out_size; (void)ws_size;
    const float* tokens = (const float*)d_in[0];
    const float* W      = (const float*)d_in[1];

    __bf16* Wb = (__bf16*)d_ws;  // 64*4096*2 = 512 KB scratch
    const int nW = NUM_EXPERTS * INPUT_DIM;
    wcvt_kernel<<<nW / 4 / 256, 256, 0, stream>>>(W, Wb, nW);

    float* scores = (float*)d_out;                      // [16384,2] f32
    int*   idx    = (int*)d_out + NUM_TOKENS * TOPK;    // [16384,2] i32 bits

    router_topk_kernel<<<NUM_TOKENS / BM, THREADS, 0, stream>>>(
        tokens, Wb, scores, idx);
}
